// RVQ_BottleneckBlock_34213709480064
// MI455X (gfx1250) — matrix operand order repaired and checked by reading, not compiled, not run on hardware
//
#include <hip/hip_runtime.h>

typedef float  v8f   __attribute__((ext_vector_type(8)));
typedef __bf16 v16bf __attribute__((ext_vector_type(16)));

#define B_   16
#define N_   2048
#define D_   128
#define Q_   8
#define K_   1024
#define ROWS (B_ * N_)        // 32768 rows total
#define MTILE 128             // rows per workgroup
#define NWG  (ROWS / MTILE)   // 256 workgroups per stage
#define WAVES 8               // 256 threads / wave32
#define BPAD 136              // padded LDS row stride (bf16 elems): 272B/row, 16B-aligned frags

// ---- manual f32 <-> bf16 (RNE), no reliance on __bf16 arithmetic ----------
static __device__ __forceinline__ unsigned short f32_to_bf16_rne(float f) {
    unsigned int u = __float_as_uint(f);
    u += 0x7FFFu + ((u >> 16) & 1u);     // round-to-nearest-even (finite inputs)
    return (unsigned short)(u >> 16);
}
static __device__ __forceinline__ float bf16_to_f32(unsigned short s) {
    return __uint_as_float(((unsigned int)s) << 16);
}

// ---------------------------------------------------------------------------
// ||c||^2 per code row (exact f32): cbsq[q*K + k] = sum_d codebooks[q,k,d]^2
// ---------------------------------------------------------------------------
__global__ void cbsq_kernel(const float* __restrict__ cb, float* __restrict__ cbsq) {
    int r = blockIdx.x * blockDim.x + threadIdx.x;   // 0 .. Q*K-1
    const float* p = cb + (size_t)r * D_;
    float s = 0.f;
#pragma unroll 8
    for (int d = 0; d < D_; ++d) s += p[d] * p[d];
    cbsq[r] = s;
}

// ---------------------------------------------------------------------------
// Split codebooks into bf16 hi/lo planes (one-time; reused 256x per stage)
// ---------------------------------------------------------------------------
__global__ void cbhilo_kernel(const float* __restrict__ cb,
                              unsigned short* __restrict__ hi,
                              unsigned short* __restrict__ lo) {
    int i = blockIdx.x * blockDim.x + threadIdx.x;   // 0 .. Q*K*D-1
    float x = cb[i];
    unsigned short hb = f32_to_bf16_rne(x);
    unsigned short lb = f32_to_bf16_rne(x - bf16_to_f32(hb));
    hi[i] = hb;
    lo[i] = lb;
}

// ---------------------------------------------------------------------------
// One RVQ stage: dist = ||c||^2 - 2 r.c with r.c via split-bf16 WMMA
// (hi*hi + lo*hi + hi*lo), argmin over K, gather, residual update, commit.
// ---------------------------------------------------------------------------
__global__ __launch_bounds__(256)
void rvq_stage_kernel(const float*          __restrict__ resIn,
                      float*                __restrict__ resOut,
                      const float*          __restrict__ cbStage,    // [K,D] f32 (gather)
                      const unsigned short* __restrict__ cbHiStage,  // [K,D] bf16 hi
                      const unsigned short* __restrict__ cbLoStage,  // [K,D] bf16 lo
                      const float*          __restrict__ cbsqStage,  // [K]
                      float*                __restrict__ outIdx,     // [ROWS,Q] as float
                      float*                __restrict__ partial,    // [NWG] this stage
                      int stage)
{
    __shared__ unsigned short hiT[2][16 * BPAD];   // double-buffered code tile (hi)
    __shared__ unsigned short loT[2][16 * BPAD];   // double-buffered code tile (lo)
    __shared__ int   idxSh[WAVES * 16];
    __shared__ float wsum[WAVES];

    const int tid  = threadIdx.x;
    const int lane = tid & 31;
    const int w    = tid >> 5;
    const int l16  = lane & 15;
    const int half = lane >> 4;
    const int rowBase = blockIdx.x * MTILE + w * 16;   // 16 rows per wave

    // --- Build A fragments: residual rows split into bf16 hi/lo, contiguous-k 16x32
    // layout: lane m=l16, half h; chunk ch covers dims [32ch,32ch+32):
    // elem e(0..7) <- dim 32ch+16h+e ; elem 8+e <- dim 32ch+16h+8+e.
    v16bf aHi[4], aLo[4];
    {
        const float* ap = resIn + (size_t)(rowBase + l16) * D_;
#pragma unroll
        for (int ch = 0; ch < 4; ++ch) {
            const int base = 32 * ch + 16 * half;
#pragma unroll
            for (int e = 0; e < 8; ++e) {
                float x0 = ap[base + e];
                unsigned short h0 = f32_to_bf16_rne(x0);
                unsigned short l0 = f32_to_bf16_rne(x0 - bf16_to_f32(h0));
                aHi[ch][e] = __builtin_bit_cast(__bf16, h0);
                aLo[ch][e] = __builtin_bit_cast(__bf16, l0);
                float x1 = ap[base + 8 + e];
                unsigned short h1 = f32_to_bf16_rne(x1);
                unsigned short l1 = f32_to_bf16_rne(x1 - bf16_to_f32(h1));
                aHi[ch][8 + e] = __builtin_bit_cast(__bf16, h1);
                aLo[ch][8 + e] = __builtin_bit_cast(__bf16, l1);
            }
        }
    }

    float minv[8];
    int   mini[8];
#pragma unroll
    for (int v = 0; v < 8; ++v) { minv[v] = 3.0e38f; mini[v] = 0; }

    // tile staging: thread t moves one 16B (8 bf16) segment per plane
    const int code = tid >> 4;
    const int seg  = tid & 15;
    const int ldsOff = code * BPAD + seg * 8;

    uint4 rh, rl;
    {
        size_t off = (size_t)code * D_ + seg * 8;
        rh = *(const uint4*)(cbHiStage + off);
        rl = *(const uint4*)(cbLoStage + off);
    }

    for (int kt = 0; kt < K_ / 16; ++kt) {
        const int buf = kt & 1;
        *(uint4*)&hiT[buf][ldsOff] = rh;
        *(uint4*)&loT[buf][ldsOff] = rl;
        __syncthreads();                       // tile[buf] ready; prev readers done

        if (kt + 1 < K_ / 16) {                // prefetch next tile into registers
            size_t off = (size_t)((kt + 1) * 16 + code) * D_ + seg * 8;
            rh = *(const uint4*)(cbHiStage + off);
            rl = *(const uint4*)(cbLoStage + off);
        }
        float cq = cbsqStage[kt * 16 + l16];

        // 12 bf16 WMMAs on 3 independent accumulator chains
        v8f c0 = {0.f,0.f,0.f,0.f,0.f,0.f,0.f,0.f};
        v8f c1 = c0, c2 = c0;
        const unsigned short* bh0 = &hiT[buf][l16 * BPAD + 16 * half];
        const unsigned short* bl0 = &loT[buf][l16 * BPAD + 16 * half];
#pragma unroll
        for (int ch = 0; ch < 4; ++ch) {
            v16bf bh = *(const v16bf*)(bh0 + 32 * ch);
            v16bf bl = *(const v16bf*)(bl0 + 32 * ch);
            c0 = __builtin_amdgcn_wmma_f32_16x16x32_bf16(false, aHi[ch], false, bh,
                                                         (short)0, c0, false, false);
            c1 = __builtin_amdgcn_wmma_f32_16x16x32_bf16(false, aLo[ch], false, bh,
                                                         (short)0, c1, false, false);
            c2 = __builtin_amdgcn_wmma_f32_16x16x32_bf16(false, aHi[ch], false, bl,
                                                         (short)0, c2, false, false);
        }

        // C layout: lane l, slot v -> row m = v + 8*(l>=16), col n = l%16
        int code0 = kt * 16 + l16;
#pragma unroll
        for (int v = 0; v < 8; ++v) {
            float dv = cq - 2.0f * (c0[v] + c1[v] + c2[v]);
            if (dv < minv[v]) { minv[v] = dv; mini[v] = code0; }  // '<' keeps lowest idx
        }
    }

    // --- cross-lane argmin over the 16 lanes of each half
#pragma unroll
    for (int v = 0; v < 8; ++v) {
#pragma unroll
        for (int mask = 1; mask < 16; mask <<= 1) {
            float ov = __shfl_xor(minv[v], mask, 32);
            int   oi = __shfl_xor(mini[v], mask, 32);
            if (ov < minv[v] || (ov == minv[v] && oi < mini[v])) { minv[v] = ov; mini[v] = oi; }
        }
    }
    if (l16 == 0) {  // lanes 0 (rows 0-7) and 16 (rows 8-15) publish
#pragma unroll
        for (int v = 0; v < 8; ++v) {
            int m = v + 8 * half;
            idxSh[w * 16 + m] = mini[v];
            int R = rowBase + m;
            outIdx[(size_t)R * Q_ + stage] = (float)mini[v];
        }
    }
    __syncthreads();

    // --- exact f32 gather + residual update + commit partial
    float acc = 0.f;
    const int d0 = lane * 4;
#pragma unroll 4
    for (int r = 0; r < 16; ++r) {
        int idxr = idxSh[w * 16 + r];
        const float4 cv = *(const float4*)(cbStage + (size_t)idxr * D_ + d0);
        int R = rowBase + r;
        float4 ro = *(const float4*)(resIn + (size_t)R * D_ + d0);
        float4 rn;
        rn.x = ro.x - cv.x; rn.y = ro.y - cv.y;
        rn.z = ro.z - cv.z; rn.w = ro.w - cv.w;
        *(float4*)(resOut + (size_t)R * D_ + d0) = rn;
        acc += rn.x * rn.x + rn.y * rn.y + rn.z * rn.z + rn.w * rn.w;
    }
#pragma unroll
    for (int mask = 1; mask < 32; mask <<= 1)
        acc += __shfl_xor(acc, mask, 32);
    if (lane == 0) wsum[w] = acc;
    __syncthreads();
    if (tid == 0) {
        float s = 0.f;
#pragma unroll
        for (int i = 0; i < WAVES; ++i) s += wsum[i];
        partial[blockIdx.x] = s;   // deterministic fixed-order sums only
    }
}

// ---------------------------------------------------------------------------
// quantized = x - final_residual (sum of selected codes telescopes)
// ---------------------------------------------------------------------------
__global__ void quant_kernel(const float* __restrict__ x,
                             const float* __restrict__ res,
                             float* __restrict__ qout)
{
    int i = blockIdx.x * blockDim.x + threadIdx.x;   // float4 index
    float4 xv = ((const float4*)x)[i];
    float4 rv = ((const float4*)res)[i];
    float4 o;
    o.x = xv.x - rv.x; o.y = xv.y - rv.y;
    o.z = xv.z - rv.z; o.w = xv.w - rv.w;
    ((float4*)qout)[i] = o;
}

// ---------------------------------------------------------------------------
// commits[q] = (sum of per-WG partials, fixed order) / (B*N*D)
// ---------------------------------------------------------------------------
__global__ void commit_kernel(const float* __restrict__ partial,
                              float* __restrict__ outCommit)
{
    int q = threadIdx.x;
    if (q < Q_) {
        float s = 0.f;
        for (int i = 0; i < NWG; ++i) s += partial[(size_t)q * NWG + i];
        outCommit[q] = s * (1.0f / ((float)ROWS * (float)D_));
    }
}

// ---------------------------------------------------------------------------
extern "C" void kernel_launch(void* const* d_in, const int* in_sizes, int n_in,
                              void* d_out, int out_size, void* d_ws, size_t ws_size,
                              hipStream_t stream)
{
    (void)in_sizes; (void)n_in; (void)out_size; (void)ws_size;
    const float* x  = (const float*)d_in[0];   // [B,N,D]
    const float* cb = (const float*)d_in[1];   // [Q,K,D]

    float* out       = (float*)d_out;
    float* outIdx    = out;                                   // B*N*Q
    float* outQuant  = out + (size_t)ROWS * Q_;               // B*N*D
    float* outCommit = outQuant + (size_t)ROWS * D_;          // Q

    float* res     = (float*)d_ws;                            // ROWS*D f32 (16 MB)
    float* cbsq    = res + (size_t)ROWS * D_;                 // Q*K f32
    float* partial = cbsq + (size_t)Q_ * K_;                  // Q*NWG f32
    unsigned short* cbHi = (unsigned short*)(partial + (size_t)Q_ * NWG); // Q*K*D bf16
    unsigned short* cbLo = cbHi + (size_t)Q_ * K_ * D_;                   // Q*K*D bf16

    cbsq_kernel<<<(Q_ * K_) / 256, 256, 0, stream>>>(cb, cbsq);
    cbhilo_kernel<<<(Q_ * K_ * D_) / 256, 256, 0, stream>>>(cb, cbHi, cbLo);

    for (int q = 0; q < Q_; ++q) {
        const float* rin = (q == 0) ? x : res;
        rvq_stage_kernel<<<NWG, 256, 0, stream>>>(
            rin, res,
            cb   + (size_t)q * K_ * D_,
            cbHi + (size_t)q * K_ * D_,
            cbLo + (size_t)q * K_ * D_,
            cbsq + (size_t)q * K_,
            outIdx,
            partial + (size_t)q * NWG,
            q);
    }

    quant_kernel<<<(ROWS * D_ / 4) / 256, 256, 0, stream>>>(x, res, outQuant);
    commit_kernel<<<1, 32, 0, stream>>>(partial, outCommit);
}
